// EdgeGate_644245095026
// MI455X (gfx1250) — matrix operand order repaired and checked by reading, not compiled, not run on hardware
//
#include <hip/hip_runtime.h>
#include <stdint.h>

typedef __attribute__((ext_vector_type(16))) __bf16 v16bf;
typedef __attribute__((ext_vector_type(8)))  float  v8f;

#define IN_DIM   384
#define HID_DIM  256
#define N_TILES  16   // 256 / 16
#define K_STEPS  12   // 384 / 32
// packed B: [k_step][n_tile][lane][8 dwords] = 12*16*32*8 dwords = 49152 dwords = 192 KB
#define B_DWORDS (K_STEPS * N_TILES * 32 * 8)

__device__ __forceinline__ uint32_t pkbf(float a, float b) {
    // pack two f32 -> two bf16 (round-to-nearest-even)
    uint32_t ua = __builtin_bit_cast(uint32_t, a);
    uint32_t ub = __builtin_bit_cast(uint32_t, b);
    ua += 0x7FFFu + ((ua >> 16) & 1u);
    ub += 0x7FFFu + ((ub >> 16) & 1u);
    return (ua >> 16) | (ub & 0xFFFF0000u);
}

// ---------------------------------------------------------------------------
// Prep: convert W1 (384x256 f32, row-major) into bf16 packed in the WMMA
// B-operand per-lane layout, ordered [k_step][n_tile][lane][v] so the main
// kernel's 32 per-K ds_load_b128s use immediate offsets.
//   lane<16 : K = 2v, 2v+1      (N = lane)
//   lane>=16: K = 2v+16, 2v+17  (N = lane-16)
// ---------------------------------------------------------------------------
__global__ void prep_W1_kernel(const float* __restrict__ W1, uint32_t* __restrict__ Bpk) {
    int idx = blockIdx.x * blockDim.x + threadIdx.x;
    if (idx >= B_DWORDS) return;
    int v    = idx & 7;
    int lane = (idx >> 3) & 31;
    int nt   = (idx >> 8) & 15;
    int ks   = idx / (8 * 32 * 16);
    int n    = nt * 16 + (lane & 15);
    int kb   = (lane < 16) ? 0 : 16;
    int k    = ks * 32 + kb + 2 * v;
    Bpk[idx] = pkbf(W1[(size_t)k * HID_DIM + n], W1[(size_t)(k + 1) * HID_DIM + n]);
}

// ---------------------------------------------------------------------------
// Main: one wave per 16-edge tile. K-outer / N-inner with 16 resident f32
// accumulators (128 VGPRs): each K-step gathers one 8-VGPR bf16 A slice, then
// issues 16 INDEPENDENT v_wmma_f32_16x16x32_bf16 with double-buffered B from
// LDS. amdgpu_waves_per_eu(1) raises the VGPR budget so nothing spills — the
// 192 KB LDS already limits the WGP to one workgroup, so register occupancy
// beyond 2 waves/SIMD is moot.
// ---------------------------------------------------------------------------
__global__ void
__attribute__((amdgpu_flat_work_group_size(256, 256), amdgpu_waves_per_eu(1)))
edge_gate_kernel(const float* __restrict__ instr,
                 const float* __restrict__ x,
                 const int*   __restrict__ ei,
                 const int*   __restrict__ batch,
                 const float* __restrict__ b1,
                 const float* __restrict__ W2,
                 const float* __restrict__ b2,
                 const uint32_t* __restrict__ Bpk,
                 float* __restrict__ out,
                 int E, int tiles) {
    extern __shared__ uint32_t Bsh[];   // B_DWORDS dwords = 192 KB dynamic LDS

    const int tid  = threadIdx.x;
    const int lane = tid & 31;
    const int wave = tid >> 5;

    // ---- cooperative stage of packed W1 into LDS (vectorized b128) ----
    {
        const uint4* g4 = reinterpret_cast<const uint4*>(Bpk);
        uint4*       s4 = reinterpret_cast<uint4*>(Bsh);
        for (int i = tid; i < B_DWORDS / 4; i += blockDim.x) s4[i] = g4[i];
    }
    __syncthreads();

    const float b2v = b2[0];
    const int row   = lane & 15;
    const int kb    = (lane < 16) ? 0 : 16;  // A-operand K sub-offset (contiguous K = 16h + s)
    const uint4* Bsh4 = reinterpret_cast<const uint4*>(Bsh);

    for (int tile = blockIdx.x * 8 + wave; tile < tiles; tile += gridDim.x * 8) {
        // ---- gather per-row (per-edge) feature base pointers ----
        int e = tile * 16 + row;
        if (e >= E) e = E - 1;               // clamp (loads only; stores guarded)
        const int src = ei[e];
        const int dst = ei[E + e];
        const int g   = batch[src];

        const float* pa = x + (size_t)src * 128;
        const float* pb = x + (size_t)dst * 128;
        const float* pu = instr + (size_t)g * 128;

        // ---- 16 resident accumulators (one per N-tile) ----
        v8f acc[N_TILES];
        #pragma unroll
        for (int nt = 0; nt < N_TILES; ++nt) acc[nt] = {};

        // ---- K-outer loop: build one A slice, fire 16 independent WMMAs ----
        for (int k = 0; k < K_STEPS; ++k) {
            // segment select: k 0-3 -> x[src], 4-7 -> x[dst], 8-11 -> instr[g]
            const float* P0 = (k < 4) ? pa : ((k < 8) ? pb : pu);
            const float* P  = P0 + (k & 3) * 32 + kb;

            // A layout: M = lane%16, K = kb+2v, kb+2v+1 (v = 0..7), matching packed B
            union AU { uint32_t u[8]; v16bf v; } a;
            float4 c0 = *reinterpret_cast<const float4*>(P);
            float4 c1 = *reinterpret_cast<const float4*>(P + 4);
            float4 c2 = *reinterpret_cast<const float4*>(P + 8);
            float4 c3 = *reinterpret_cast<const float4*>(P + 12);
            a.u[0] = pkbf(c0.x, c0.y); a.u[1] = pkbf(c0.z, c0.w);
            a.u[2] = pkbf(c1.x, c1.y); a.u[3] = pkbf(c1.z, c1.w);
            a.u[4] = pkbf(c2.x, c2.y); a.u[5] = pkbf(c2.z, c2.w);
            a.u[6] = pkbf(c3.x, c3.y); a.u[7] = pkbf(c3.z, c3.w);

            // per-k LDS base (uint4 units); nt stride = 1024 B -> imm offsets
            const int base4 = (k * N_TILES * 32 + lane) * 2;

            // double-buffered B: load nt+1 before issuing WMMA nt
            union BU { uint4 q[2]; v16bf v; } bu0, bu1;
            bu0.q[0] = Bsh4[base4];
            bu0.q[1] = Bsh4[base4 + 1];
            #pragma unroll
            for (int nt = 0; nt < N_TILES; ++nt) {
                if (nt + 1 < N_TILES) {
                    bu1.q[0] = Bsh4[base4 + (nt + 1) * 64];
                    bu1.q[1] = Bsh4[base4 + (nt + 1) * 64 + 1];
                }
                acc[nt] = __builtin_amdgcn_wmma_f32_16x16x32_bf16(
                    /*neg_a=*/false, a.v, /*neg_b=*/false, bu0.v,
                    /*c_mod=*/(short)0, acc[nt], /*reuse_a=*/false, /*reuse_b=*/false);
                bu0 = bu1;
            }
        }

        // ---- bias + SiLU + W2 fold, per N-tile ----
        float partial[8];
        #pragma unroll
        for (int d = 0; d < 8; ++d) partial[d] = 0.0f;

        #pragma unroll
        for (int nt = 0; nt < N_TILES; ++nt) {
            const int col   = nt * 16 + row;
            const float b1v = b1[col];
            const float w2v = W2[col];
            #pragma unroll
            for (int d = 0; d < 8; ++d) {
                float h  = acc[nt][d] + b1v;
                float sv = h / (1.0f + __expf(-h));   // SiLU
                partial[d] += sv * w2v;
            }
        }

        // ---- reduce across 16 lanes (rows 0-7 in lanes 0-15, 8-15 in 16-31) ----
        #pragma unroll
        for (int d = 0; d < 8; ++d) {
            float p = partial[d];
            p += __shfl_xor(p, 1, 16);
            p += __shfl_xor(p, 2, 16);
            p += __shfl_xor(p, 4, 16);
            p += __shfl_xor(p, 8, 16);
            partial[d] = p;
        }
        if ((lane & 15) == 0) {
            const int rowbase = (lane >> 4) * 8;   // lane 0 -> rows 0-7, lane 16 -> rows 8-15
            #pragma unroll
            for (int d = 0; d < 8; ++d) {
                int eo = tile * 16 + rowbase + d;
                if (eo < E) out[eo] = partial[d] + b2v;
            }
        }
    }
}

extern "C" void kernel_launch(void* const* d_in, const int* in_sizes, int n_in,
                              void* d_out, int out_size, void* d_ws, size_t ws_size,
                              hipStream_t stream) {
    const float* instr = (const float*)d_in[0];
    const float* x     = (const float*)d_in[1];
    const int*   ei    = (const int*)  d_in[2];
    const int*   batch = (const int*)  d_in[3];
    const float* W1    = (const float*)d_in[4];
    const float* b1    = (const float*)d_in[5];
    const float* W2    = (const float*)d_in[6];
    const float* b2    = (const float*)d_in[7];
    float*       out   = (float*)d_out;

    const int E     = in_sizes[2] / 2;
    const int tiles = (E + 15) / 16;

    uint32_t* Bpk = (uint32_t*)d_ws;

    // 1) pack W1 into WMMA B-operand bf16 layout ([k][nt][lane][v])
    prep_W1_kernel<<<(B_DWORDS + 255) / 256, 256, 0, stream>>>(W1, Bpk);

    // 2) main fused gather + GEMM + SiLU + reduce kernel
    const int blocks = (tiles + 7) / 8;          // 8 waves (tiles) per workgroup
    const size_t lds = (size_t)B_DWORDS * 4;     // 192 KB dynamic LDS
    edge_gate_kernel<<<blocks, 256, lds, stream>>>(instr, x, ei, batch,
                                                   b1, W2, b2, Bpk, out, E, tiles);
}
